// MLPMambaBlock_35192962023485
// MI455X (gfx1250) — compile-verified, matrix operand order checked by reading, not run on hardware
//
#include <hip/hip_runtime.h>

typedef __bf16 bf16_t;
typedef __attribute__((ext_vector_type(16))) __bf16 v16bf;
typedef __attribute__((ext_vector_type(8)))  float  v8f;
typedef int v4i __attribute__((vector_size(16)));   // matches builtin's int __vector(4)

#define GLOBAL_AS __attribute__((address_space(1)))
#define LDS_AS    __attribute__((address_space(3)))

#if __has_builtin(__builtin_amdgcn_global_load_async_to_lds_b128) && \
    __has_builtin(__builtin_amdgcn_s_wait_asynccnt)
#define USE_ASYNC_LDS 1
#else
#define USE_ASYNC_LDS 0
#endif

// ---------------- problem constants ----------------
constexpr int DM   = 2048;   // D_MODEL
constexpr int DI   = 4096;   // D_INNER
constexpr int DST  = 16;     // D_STATE
constexpr int DCONV= 4;      // D_CONV
constexpr int DTR  = 128;    // DT_RANK
constexpr int HID  = 8192;   // HIDDEN
constexpr int Bb   = 2;
constexpr int Ls   = 2048;
constexpr int MR   = Bb * Ls;          // 4096 rows
constexpr int NX   = DTR + 2 * DST;    // 160

// ---------------- helpers ----------------
__device__ __forceinline__ bf16_t f2bf(float f) {
  union { float f; unsigned u; } uf; uf.f = f;
  unsigned u = uf.u;
  unsigned r = u + 0x7FFFu + ((u >> 16) & 1u);   // round-to-nearest-even
  unsigned short h = (unsigned short)(r >> 16);
  union { unsigned short s; bf16_t b; } ub; ub.s = h;
  return ub.b;
}
__device__ __forceinline__ float siluf(float v) {
  return v * (1.0f / (1.0f + __expf(-v)));
}

// ---------------- fp32 -> bf16 convert ----------------
__global__ __launch_bounds__(256)
void f32_to_bf16(const float* __restrict__ in, bf16_t* __restrict__ out, size_t n) {
  size_t i = (size_t)blockIdx.x * 256 + threadIdx.x;
  if (i < n) out[i] = f2bf(in[i]);
}

// ---------------- LayerNorm -> bf16 ----------------
__global__ __launch_bounds__(256)
void ln_kernel(const float* __restrict__ x, const float* __restrict__ w,
               const float* __restrict__ b, bf16_t* __restrict__ o, int D) {
  __shared__ float sh[512];
  const int row = blockIdx.x, tid = threadIdx.x;
  const float* xr = x + (size_t)row * D;
  float s = 0.f, ss = 0.f;
  for (int i = tid; i < D; i += 256) { float v = xr[i]; s += v; ss += v * v; }
  sh[tid] = s; sh[256 + tid] = ss; __syncthreads();
  for (int k = 128; k > 0; k >>= 1) {
    if (tid < k) { sh[tid] += sh[tid + k]; sh[256 + tid] += sh[256 + tid + k]; }
    __syncthreads();
  }
  float mean = sh[0] / (float)D;
  float var  = sh[256] / (float)D - mean * mean;
  float rstd = rsqrtf(var + 1e-5f);
  for (int i = tid; i < D; i += 256)
    o[(size_t)row * D + i] = f2bf((xr[i] - mean) * rstd * w[i] + b[i]);
}

// residual add + LayerNorm: xres = x + mix ; o = LN(xres) in bf16
__global__ __launch_bounds__(256)
void resid_ln_kernel(const float* __restrict__ x, const float* __restrict__ mix,
                     const float* __restrict__ w, const float* __restrict__ b,
                     float* __restrict__ xres, bf16_t* __restrict__ o, int D) {
  __shared__ float sh[512];
  const int row = blockIdx.x, tid = threadIdx.x;
  const size_t base = (size_t)row * D;
  float s = 0.f, ss = 0.f;
  for (int i = tid; i < D; i += 256) {
    float v = x[base + i] + mix[base + i];
    xres[base + i] = v;
    s += v; ss += v * v;
  }
  sh[tid] = s; sh[256 + tid] = ss; __syncthreads();
  for (int k = 128; k > 0; k >>= 1) {
    if (tid < k) { sh[tid] += sh[tid + k]; sh[256 + tid] += sh[256 + tid + k]; }
    __syncthreads();
  }
  float mean = sh[0] / (float)D;
  float var  = sh[256] / (float)D - mean * mean;
  float rstd = rsqrtf(var + 1e-5f);
  for (int i = tid; i < D; i += 256)
    o[base + i] = f2bf((xres[base + i] - mean) * rstd * w[i] + b[i]);
}

// ---------------- WMMA bf16 GEMM: C[M,N] = A[M,K] * W[N,K]^T ----------------
// Double-buffered LDS pipeline; async global->LDS copies (ASYNCcnt) when available.
constexpr int BM = 128, BN = 128, BK = 32;
constexpr int LDSS = 48;   // bf16 elems/row: 32 data + 16 pad -> 96B rows, 32B-aligned frags

__global__ __launch_bounds__(256)
void gemm_bf16_wmma(const bf16_t* __restrict__ A, const bf16_t* __restrict__ W,
                    float* __restrict__ C, int M, int N, int K) {
  __shared__ __align__(32) bf16_t As[2][BM * LDSS];
  __shared__ __align__(32) bf16_t Bs[2][BN * LDSS];

  const int tid  = threadIdx.x;
  const int lane = tid & 31;
  const int wave = tid >> 5;                // 8 waves
  const int wm   = (wave & 3) * 32;         // 4 waves along M
  const int wn   = (wave >> 2) * 64;        // 2 waves along N
  const int bm   = blockIdx.x * BM;
  const int bn   = blockIdx.y * BN;

  // tile loaders: each thread moves 16 bf16 (32B) per operand per K-step
  const int lrow = tid >> 1;                // 0..127
  const int lcol = (tid & 1) * 16;          // 0 or 16

  // fragment addressing: lane -> row (M for A / N for B), K half
  const int mrow = lane & 15;
  const int kh   = (lane >> 4) * 16;

  const v8f vzero = {0.f, 0.f, 0.f, 0.f, 0.f, 0.f, 0.f, 0.f};
  v8f acc[2][4];
#pragma unroll
  for (int i = 0; i < 2; ++i)
#pragma unroll
    for (int j = 0; j < 4; ++j) acc[i][j] = vzero;

  const int    nrow = bn + lrow;
  const size_t arow = (size_t)(bm + lrow) * K;
  // clamp OOB weight rows (N=160 case): garbage only reaches masked-out columns
  const size_t wrow = (size_t)(nrow < N ? nrow : N - 1) * K;

#if USE_ASYNC_LDS
  auto stage = [&](int buf, int k0) {
    const char* gA = (const char*)(A + arow + k0 + lcol);
    const char* gW = (const char*)(W + wrow + k0 + lcol);
    char* lA = (char*)(&As[buf][lrow * LDSS + lcol]);
    char* lW = (char*)(&Bs[buf][lrow * LDSS + lcol]);
    __builtin_amdgcn_global_load_async_to_lds_b128(
        (GLOBAL_AS v4i*)(uintptr_t)gA,        (LDS_AS v4i*)(unsigned)(uintptr_t)lA, 0, 0);
    __builtin_amdgcn_global_load_async_to_lds_b128(
        (GLOBAL_AS v4i*)(uintptr_t)(gA + 16), (LDS_AS v4i*)(unsigned)(uintptr_t)(lA + 16), 0, 0);
    __builtin_amdgcn_global_load_async_to_lds_b128(
        (GLOBAL_AS v4i*)(uintptr_t)gW,        (LDS_AS v4i*)(unsigned)(uintptr_t)lW, 0, 0);
    __builtin_amdgcn_global_load_async_to_lds_b128(
        (GLOBAL_AS v4i*)(uintptr_t)(gW + 16), (LDS_AS v4i*)(unsigned)(uintptr_t)(lW + 16), 0, 0);
  };
#else
  auto stage = [&](int buf, int k0) {
    v16bf av = *(const v16bf*)(A + arow + k0 + lcol);
    v16bf wv = *(const v16bf*)(W + wrow + k0 + lcol);
    *(v16bf*)(&As[buf][lrow * LDSS + lcol]) = av;
    *(v16bf*)(&Bs[buf][lrow * LDSS + lcol]) = wv;
  };
#endif

  stage(0, 0);
  int buf = 0;
  for (int k0 = 0; k0 < K; k0 += BK) {
#if USE_ASYNC_LDS
    __builtin_amdgcn_s_wait_asynccnt(0);   // my copies for buf are done
#endif
    __syncthreads();                        // everyone's copies done; prev reads done
    if (k0 + BK < K) {
      stage(buf ^ 1, k0 + BK);              // overlap next slab with this compute
      __builtin_prefetch(A + arow + k0 + 2 * BK, 0, 3);  // warm caches 2 slabs ahead
      __builtin_prefetch(W + wrow + k0 + 2 * BK, 0, 3);
    }

    const bf16_t* Ab = As[buf];
    const bf16_t* Wb = Bs[buf];
    v16bf af[2], bfr[4];
#pragma unroll
    for (int mi = 0; mi < 2; ++mi)
      af[mi] = *(const v16bf*)(Ab + (wm + mi * 16 + mrow) * LDSS + kh);
#pragma unroll
    for (int ni = 0; ni < 4; ++ni)
      bfr[ni] = *(const v16bf*)(Wb + (wn + ni * 16 + mrow) * LDSS + kh);

#pragma unroll
    for (int mi = 0; mi < 2; ++mi)
#pragma unroll
      for (int ni = 0; ni < 4; ++ni)
        acc[mi][ni] = __builtin_amdgcn_wmma_f32_16x16x32_bf16(
            false, af[mi], false, bfr[ni], (short)0, acc[mi][ni], false, false);
    buf ^= 1;
  }

  // C/D layout: VGPR r, lanes 0-15 -> M=r, lanes 16-31 -> M=r+8; N = lane&15
  const int nlo = lane & 15;
  const int mhi = (lane >> 4) * 8;
#pragma unroll
  for (int mi = 0; mi < 2; ++mi) {
#pragma unroll
    for (int ni = 0; ni < 4; ++ni) {
      const int n = bn + wn + ni * 16 + nlo;
      if (n < N) {
        const int m0 = bm + wm + mi * 16 + mhi;
#pragma unroll
        for (int r = 0; r < 8; ++r)
          C[(size_t)(m0 + r) * N + n] = acc[mi][ni][r];
      }
    }
  }
}

// ---------------- causal depthwise conv (d_conv=4) + SiLU ----------------
__global__ __launch_bounds__(256)
void conv_silu_kernel(const float* __restrict__ xz, const float* __restrict__ cw,
                      const float* __restrict__ cb, float* __restrict__ xc,
                      bf16_t* __restrict__ xcbf) {
  int idx = blockIdx.x * 256 + threadIdx.x;   // over MR*DI
  int d = idx & (DI - 1);
  int m = idx >> 12;                           // DI = 4096
  int l = m & (Ls - 1);
  float acc = cb[d];
#pragma unroll
  for (int j = 0; j < DCONV; ++j) {
    int ls = l + j - (DCONV - 1);
    if (ls >= 0) {
      int rr = m + j - (DCONV - 1);            // same batch (ls>=0 guarantees it)
      acc += xz[(size_t)rr * (2 * DI) + d] * cw[d * DCONV + j];
    }
  }
  float v = siluf(acc);
  xc[idx]   = v;
  xcbf[idx] = f2bf(v);
}

// ---------------- extract dt part of x_dbl -> bf16 ----------------
__global__ __launch_bounds__(256)
void dt_extract_kernel(const float* __restrict__ xdbl, bf16_t* __restrict__ dtbf) {
  int idx = blockIdx.x * 256 + threadIdx.x;   // over MR*DTR
  int m = idx >> 7, r = idx & (DTR - 1);
  dtbf[idx] = f2bf(xdbl[(size_t)m * NX + r]);
}

// ---------------- softplus(dt + bias) in place ----------------
__global__ __launch_bounds__(256)
void softplus_kernel(float* __restrict__ dta, const float* __restrict__ dtb) {
  int idx = blockIdx.x * 256 + threadIdx.x;   // over MR*DI
  int d = idx & (DI - 1);
  float v = dta[idx] + dtb[d];
  dta[idx] = (v > 20.f) ? v : log1pf(__expf(v));
}

// ---------------- selective scan: thread = (b, d), 16 states in VGPRs ----------------
// Software-pipelined: step t+1 operands load while step t's 16-exp chain executes.
__global__ __launch_bounds__(256)
void scan_kernel(const float* __restrict__ dta, const float* __restrict__ xc,
                 const float* __restrict__ xdbl, const float* __restrict__ A_log,
                 float* __restrict__ y) {
  const int blk = blockIdx.x;                 // Bb * (DI/256) = 32 blocks
  const int b   = blk >> 4;
  const int d   = ((blk & 15) << 8) + threadIdx.x;
  float a[DST], h[DST];
#pragma unroll
  for (int s = 0; s < DST; ++s) { a[s] = -__expf(A_log[d * DST + s]); h[s] = 0.f; }

  auto loadstep = [&](int t, float& dt_, float& xv_, float* Bv_, float* Cv_) {
    const size_t base = (size_t)b * Ls + t;
    dt_ = dta[base * DI + d];
    xv_ = xc[base * DI + d];
    const float4* q = (const float4*)(xdbl + base * NX + DTR);  // B then C, broadcast
#pragma unroll
    for (int j = 0; j < 4; ++j) {
      float4 bq = q[j], cq = q[4 + j];
      Bv_[4 * j] = bq.x; Bv_[4 * j + 1] = bq.y; Bv_[4 * j + 2] = bq.z; Bv_[4 * j + 3] = bq.w;
      Cv_[4 * j] = cq.x; Cv_[4 * j + 1] = cq.y; Cv_[4 * j + 2] = cq.z; Cv_[4 * j + 3] = cq.w;
    }
  };

  float dt, xv, Bv[DST], Cv[DST];
  loadstep(0, dt, xv, Bv, Cv);
  for (int t = 0; t < Ls; ++t) {
    float dtn = 0.f, xvn = 0.f, Bn[DST], Cn[DST];
#pragma unroll
    for (int s = 0; s < DST; ++s) { Bn[s] = 0.f; Cn[s] = 0.f; }
    if (t + 1 < Ls) loadstep(t + 1, dtn, xvn, Bn, Cn);   // issue early, overlaps exp chain

    float yv = 0.f;
#pragma unroll
    for (int s = 0; s < DST; ++s) {
      float dA = __expf(dt * a[s]);
      h[s] = h[s] * dA + dt * Bv[s] * xv;
      yv  += h[s] * Cv[s];
    }
    y[((size_t)b * Ls + t) * DI + d] = yv;

    dt = dtn; xv = xvn;
#pragma unroll
    for (int s = 0; s < DST; ++s) { Bv[s] = Bn[s]; Cv[s] = Cn[s]; }
  }
}

// ---------------- (y + xc*D) * silu(z) -> bf16 ----------------
__global__ __launch_bounds__(256)
void gate_kernel(const float* __restrict__ y, const float* __restrict__ xc,
                 const float* __restrict__ xz, const float* __restrict__ Dsk,
                 bf16_t* __restrict__ out) {
  int idx = blockIdx.x * 256 + threadIdx.x;   // over MR*DI
  int d = idx & (DI - 1);
  int m = idx >> 12;
  float z = xz[(size_t)m * (2 * DI) + DI + d];
  out[idx] = f2bf((y[idx] + xc[idx] * Dsk[d]) * siluf(z));
}

// ---------------- SwiGLU: silu(gate)*a -> bf16 ----------------
__global__ __launch_bounds__(256)
void swiglu_kernel(const float* __restrict__ h, bf16_t* __restrict__ out) {
  int idx = blockIdx.x * 256 + threadIdx.x;   // over MR*(HID/2)
  int d = idx & (HID / 2 - 1);
  int m = idx >> 12;
  float av = h[(size_t)m * HID + d];
  float g  = h[(size_t)m * HID + HID / 2 + d];
  out[idx] = f2bf(siluf(g) * av);
}

// ---------------- out += residual ----------------
__global__ __launch_bounds__(256)
void final_add_kernel(float* __restrict__ out, const float* __restrict__ xres) {
  int idx = blockIdx.x * 256 + threadIdx.x;   // over MR*DM
  out[idx] += xres[idx];
}

// ---------------- host orchestration ----------------
extern "C" void kernel_launch(void* const* d_in, const int* in_sizes, int n_in,
                              void* d_out, int out_size, void* d_ws, size_t ws_size,
                              hipStream_t stream) {
  (void)in_sizes; (void)n_in; (void)out_size; (void)ws_size;
  const float* x          = (const float*)d_in[0];
  const float* ln1_w      = (const float*)d_in[1];
  const float* ln1_b      = (const float*)d_in[2];
  const float* ln2_w      = (const float*)d_in[3];
  const float* ln2_b      = (const float*)d_in[4];
  const float* in_proj_w  = (const float*)d_in[5];
  const float* conv_w     = (const float*)d_in[6];
  const float* conv_b     = (const float*)d_in[7];
  const float* x_proj_w   = (const float*)d_in[8];
  const float* dt_proj_w  = (const float*)d_in[9];
  const float* dt_proj_b  = (const float*)d_in[10];
  const float* A_log      = (const float*)d_in[11];
  const float* D_skip     = (const float*)d_in[12];
  const float* out_proj_w = (const float*)d_in[13];
  const float* ff_proj_w  = (const float*)d_in[14];
  const float* ff_out_w   = (const float*)d_in[15];
  float* out = (float*)d_out;

  // bump allocator over workspace (256B aligned)
  char* base = (char*)d_ws;
  size_t off = 0;
  auto alloc = [&](size_t bytes) -> char* {
    char* p = base + off;
    off = (off + bytes + 255) & ~(size_t)255;
    return p;
  };
  bf16_t* w_in  = (bf16_t*)alloc((size_t)2 * DI * DM * 2);
  bf16_t* w_xp  = (bf16_t*)alloc((size_t)NX * DI * 2);
  bf16_t* w_dt  = (bf16_t*)alloc((size_t)DI * DTR * 2);
  bf16_t* w_op  = (bf16_t*)alloc((size_t)DM * DI * 2);
  bf16_t* w_fp  = (bf16_t*)alloc((size_t)HID * DM * 2);
  bf16_t* w_fo  = (bf16_t*)alloc((size_t)DM * (HID / 2) * 2);
  bf16_t* ln1bf = (bf16_t*)alloc((size_t)MR * DM * 2);
  bf16_t* ln2bf = (bf16_t*)alloc((size_t)MR * DM * 2);
  float*  xz    = (float*)alloc((size_t)MR * 2 * DI * 4);   // reused as ff hidden
  float*  xc    = (float*)alloc((size_t)MR * DI * 4);
  bf16_t* xcbf  = (bf16_t*)alloc((size_t)MR * DI * 2);
  float*  xdbl  = (float*)alloc((size_t)MR * NX * 4);
  bf16_t* dtbf  = (bf16_t*)alloc((size_t)MR * DTR * 2);
  float*  dta   = (float*)alloc((size_t)MR * DI * 4);
  float*  yscan = (float*)alloc((size_t)MR * DI * 4);
  bf16_t* ygbf  = (bf16_t*)alloc((size_t)MR * DI * 2);      // reused as swiglu out
  float*  mix   = (float*)alloc((size_t)MR * DM * 4);
  float*  xres  = (float*)alloc((size_t)MR * DM * 4);

  auto cvt = [&](const float* s, bf16_t* d, size_t n) {
    f32_to_bf16<<<dim3((unsigned)((n + 255) / 256)), dim3(256), 0, stream>>>(s, d, n);
  };
  // weights -> bf16 once; ~98MB total, fully L2-resident afterwards
  cvt(in_proj_w,  w_in, (size_t)2 * DI * DM);
  cvt(x_proj_w,   w_xp, (size_t)NX * DI);
  cvt(dt_proj_w,  w_dt, (size_t)DI * DTR);
  cvt(out_proj_w, w_op, (size_t)DM * DI);
  cvt(ff_proj_w,  w_fp, (size_t)HID * DM);
  cvt(ff_out_w,   w_fo, (size_t)DM * (HID / 2));

  auto gemm = [&](const bf16_t* A, const bf16_t* W, float* C, int M, int N, int K) {
    gemm_bf16_wmma<<<dim3(M / BM, (N + BN - 1) / BN), dim3(256), 0, stream>>>(A, W, C, M, N, K);
  };

  ln_kernel<<<MR, 256, 0, stream>>>(x, ln1_w, ln1_b, ln1bf, DM);
  gemm(ln1bf, w_in, xz, MR, 2 * DI, DM);                               // in_proj
  conv_silu_kernel<<<(MR * DI) / 256, 256, 0, stream>>>(xz, conv_w, conv_b, xc, xcbf);
  gemm(xcbf, w_xp, xdbl, MR, NX, DI);                                  // x_proj (N=160)
  dt_extract_kernel<<<(MR * DTR) / 256, 256, 0, stream>>>(xdbl, dtbf);
  gemm(dtbf, w_dt, dta, MR, DI, DTR);                                  // dt_proj
  softplus_kernel<<<(MR * DI) / 256, 256, 0, stream>>>(dta, dt_proj_b);
  scan_kernel<<<Bb * (DI / 256), 256, 0, stream>>>(dta, xc, xdbl, A_log, yscan);
  gate_kernel<<<(MR * DI) / 256, 256, 0, stream>>>(yscan, xc, xz, D_skip, ygbf);
  gemm(ygbf, w_op, mix, MR, DM, DI);                                   // out_proj
  resid_ln_kernel<<<MR, 256, 0, stream>>>(x, mix, ln2_w, ln2_b, xres, ln2bf, DM);
  float* hbuf = xz;                                                    // reuse big buffer
  gemm(ln2bf, w_fp, hbuf, MR, HID, DM);                                // ff_proj
  swiglu_kernel<<<(MR * (HID / 2)) / 256, 256, 0, stream>>>(hbuf, ygbf);
  gemm(ygbf, w_fo, out, MR, DM, HID / 2);                              // ff_out
  final_add_kernel<<<(MR * DM) / 256, 256, 0, stream>>>(out, xres);
}
